// CatEmbedder_90701119357682
// MI455X (gfx1250) — matrix operand order repaired and checked by reading, not compiled, not run on hardware
//
#include <hip/hip_runtime.h>

typedef __attribute__((ext_vector_type(16))) _Float16 v16h;
typedef __attribute__((ext_vector_type(8)))  _Float16 v8h;
typedef __attribute__((ext_vector_type(4)))  _Float16 v4h;
typedef __attribute__((ext_vector_type(8)))  float    v8f;
typedef __attribute__((ext_vector_type(4)))  int      v4i;

#define L_CATS 128
#define D_EMB  64
#define STR    72                 // padded LDS row stride (f16 elems): 144B, 16B aligned
#define INV_NORM (1.0f / 129.0f)  // 1/(L + PROBE)
#define WELEMS  (3 * D_EMB * STR)            // f16 elements in weight image
#define WBYTES  (WELEMS * 2)                 // 27648 bytes

#define GAS __attribute__((address_space(1)))
#define LAS __attribute__((address_space(3)))

struct __align__(16) Smem {
  _Float16 X0[L_CATS * STR];   // 18432 B (off 0)
  _Float16 X1[L_CATS * STR];   // 18432 B (off 18432)
  _Float16 Wacc[D_EMB * STR];  // acc_w transposed [e][d]   (off 36864, 16B aligned)
  _Float16 Wg0[D_EMB * STR];   // gw[0] row-major [e][d]
  _Float16 Wg1[D_EMB * STR];   // gw[1] row-major [e][d]
  float S[D_EMB];              // colsum(emb) @ acc_w
  float sSum[D_EMB];
  float sQ[D_EMB];
  float loc0[D_EMB];
  float loc1[D_EMB];
  float loc2[D_EMB];
  float accB[D_EMB];
  float gB0[D_EMB];
  float gB1[D_EMB];
};

// ---- One-shot weight prep: f32 -> f16, transpose acc_w, pad stride to 72 ----
__global__ void prep_weights(const float* __restrict__ acc_w,
                             const float* __restrict__ gw,
                             _Float16* __restrict__ wsf) {
  const int i = blockIdx.x * 256 + threadIdx.x;   // 0..12287
  if (i >= 3 * D_EMB * D_EMB) return;
  const int m = i >> 12;          // matrix id
  const int r = (i >> 6) & 63;    // e
  const int c = i & 63;           // d
  const float v = (m == 0) ? acc_w[c * D_EMB + r]
                           : gw[(m - 1) * D_EMB * D_EMB + r * D_EMB + c];
  wsf[m * D_EMB * STR + r * STR + c] = (_Float16)v;
}

// One wave computes a 16x64 tile of Y = f(X[128x64] @ W-tiles).
// MODE 0: adj/bias/relu -> LDS.  MODE 1: bias/relu -> LDS.
// MODE 2: bias + 0.5*g + 0.5*loc -> global f32 (nontemporal).
template <int MODE>
__device__ __forceinline__ void stage_mm(const _Float16* __restrict__ X,
                                         const _Float16* __restrict__ W,
                                         _Float16* __restrict__ Y,
                                         const float* __restrict__ bias,
                                         const float* __restrict__ Svec,
                                         const float* __restrict__ locv,
                                         float* __restrict__ gout,
                                         int mt, int lane) {
  const int lr = lane & 15;
  const int hs = lane >> 4;
  v8f acc[4] = {v8f{}, v8f{}, v8f{}, v8f{}};

#pragma unroll
  for (int kt = 0; kt < 2; ++kt) {
    const _Float16* ap = X + (mt * 16 + lr) * STR + kt * 32;
    v8h a0 = *(const v8h*)(ap + hs * 8);
    v8h a1 = *(const v8h*)(ap + 16 + hs * 8);
    v16h a = __builtin_shufflevector(a0, a1, 0, 1, 2, 3, 4, 5, 6, 7,
                                     8, 9, 10, 11, 12, 13, 14, 15);
#pragma unroll
    for (int nt = 0; nt < 4; ++nt) {
      const _Float16* bp = W + (nt * 16 + lr) * STR + kt * 32;
      v8h b0 = *(const v8h*)(bp + hs * 8);
      v8h b1 = *(const v8h*)(bp + 16 + hs * 8);
      v16h bf = __builtin_shufflevector(b0, b1, 0, 1, 2, 3, 4, 5, 6, 7,
                                        8, 9, 10, 11, 12, 13, 14, 15);
      acc[nt] = __builtin_amdgcn_wmma_f32_16x16x32_f16(
          false, a, false, bf, (short)0, acc[nt], false, false);
    }
  }

#pragma unroll
  for (int nt = 0; nt < 4; ++nt) {
    const int col = nt * 16 + lr;
#pragma unroll
    for (int r = 0; r < 8; ++r) {
      const int row = mt * 16 + hs * 8 + r;
      float v = acc[nt][r];
      if (MODE == 0) {
        v = (v + Svec[col]) * INV_NORM + bias[col];
        Y[row * STR + col] = (_Float16)fmaxf(v, 0.0f);
      } else if (MODE == 1) {
        Y[row * STR + col] = (_Float16)fmaxf(v + bias[col], 0.0f);
      } else {
        v = v + bias[col];
        __builtin_nontemporal_store(0.5f * v + 0.5f * locv[col],
                                    &gout[(size_t)row * D_EMB + col]);
      }
    }
  }
}

template <bool PRECONV>
__global__ __launch_bounds__(256, 2)
void catembed_fused(const int* __restrict__ cat,
                    const float* __restrict__ emb_table,
                    const float* __restrict__ acc_w,
                    const float* __restrict__ acc_b,
                    const float* __restrict__ gw,
                    const float* __restrict__ gb,
                    const float* __restrict__ lw,
                    const float* __restrict__ lb,
                    const _Float16* __restrict__ wsf,
                    float* __restrict__ out) {
  __shared__ Smem sm;
  const int b = blockIdx.x;
  const int tid = threadIdx.x;

  // ---- Weights into LDS ----
  if (PRECONV) {
#if __has_builtin(__builtin_amdgcn_global_load_async_to_lds_b128)
    {
      const char* gsrc = (const char*)wsf;
      char* ldst = (char*)sm.Wacc;
      for (int off = tid * 16; off < WBYTES; off += 256 * 16) {
        __builtin_amdgcn_global_load_async_to_lds_b128(
            (GAS v4i*)(gsrc + off), (LAS v4i*)(ldst + off), 0, 0);
      }
    }
#else
    for (int i = tid; i < WBYTES / 16; i += 256)
      ((int4*)sm.Wacc)[i] = ((const int4*)wsf)[i];
#endif
  } else {
    for (int i = tid; i < D_EMB * D_EMB; i += 256) {
      const int e = i >> 6, d = i & 63;
      sm.Wacc[e * STR + d] = (_Float16)acc_w[d * D_EMB + e];
      sm.Wg0[e * STR + d]  = (_Float16)gw[e * D_EMB + d];
      sm.Wg1[e * STR + d]  = (_Float16)gw[D_EMB * D_EMB + e * D_EMB + d];
    }
  }

  // ---- Biases + zero reduction arrays ----
  if (tid < D_EMB) {
    sm.accB[tid] = acc_b[tid];
    sm.gB0[tid]  = gb[tid];
    sm.gB1[tid]  = gb[D_EMB + tid];
    sm.sSum[tid] = 0.0f;
    sm.sQ[tid]   = 0.0f;
  }

  // ---- Gather embedding rows: 2 threads/row, vectorized ----
  {
    const int l = tid >> 1, h = tid & 1;
    const int idx = cat[b * L_CATS + l];
    const float4* src = (const float4*)(emb_table + idx * D_EMB + h * 32);
    _Float16* dst = sm.X0 + l * STR + h * 32;
#pragma unroll
    for (int j = 0; j < 8; ++j) {
      const float4 v = src[j];
      v4h p = {(_Float16)v.x, (_Float16)v.y, (_Float16)v.z, (_Float16)v.w};
      *(v4h*)(dst + j * 4) = p;
    }
  }

  if (PRECONV) {
#if __has_builtin(__builtin_amdgcn_global_load_async_to_lds_b128)
#if __has_builtin(__builtin_amdgcn_s_wait_asynccnt)
    __builtin_amdgcn_s_wait_asynccnt(0);
#else
    asm volatile("s_wait_asynccnt 0" ::: "memory");
#endif
#endif
  }
  __syncthreads();

  // ---- Column sums: 4-way split over rows, ds_add_f32 reduction ----
  {
    const int col = tid & 63;
    const int g = tid >> 6;
    float s = 0.0f, q = 0.0f;
#pragma unroll 4
    for (int l = g * 32; l < g * 32 + 32; ++l) {
      const float x = (float)sm.X0[l * STR + col];
      s += x; q += x * x;
    }
    atomicAdd(&sm.sSum[col], s);
    atomicAdd(&sm.sQ[col], q);
  }
  __syncthreads();

  // ---- S[e] = s . acc_w[:,e] ; FM seed loc0 ----
  if (tid < D_EMB) {
    float a = 0.0f;
    for (int d = 0; d < D_EMB; ++d) a += sm.sSum[d] * acc_w[d * D_EMB + tid];
    sm.S[tid] = a;
  } else if (tid < 2 * D_EMB) {
    const int e = tid - D_EMB;
    const float s = sm.sSum[e];
    sm.loc0[e] = 0.5f * (s * s - sm.sQ[e]);
  }
  __syncthreads();

  // ---- FM layer 0 (relu) ----
  if (tid < D_EMB) {
    float a = lb[tid];
    for (int d = 0; d < D_EMB; ++d) a += sm.loc0[d] * lw[tid * D_EMB + d];
    sm.loc1[tid] = fmaxf(a, 0.0f);
  }
  __syncthreads();

  // ---- FM layer 1 (no relu) ----
  if (tid < D_EMB) {
    float a = lb[D_EMB + tid];
    for (int d = 0; d < D_EMB; ++d)
      a += sm.loc1[d] * lw[D_EMB * D_EMB + tid * D_EMB + d];
    sm.loc2[tid] = a;
  }
  __syncthreads();

  // ---- Three fused 128x64x64 GEMMs on WMMA ----
  const int mt = tid >> 5;
  const int lane = tid & 31;
  stage_mm<0>(sm.X0, sm.Wacc, sm.X1, sm.accB, sm.S, nullptr, nullptr, mt, lane);
  __syncthreads();
  stage_mm<1>(sm.X1, sm.Wg0, sm.X0, sm.gB0, nullptr, nullptr, nullptr, mt, lane);
  __syncthreads();
  float* gout = out + (size_t)b * L_CATS * D_EMB;
  stage_mm<2>(sm.X0, sm.Wg1, nullptr, sm.gB1, nullptr, sm.loc2, gout, mt, lane);
}

extern "C" void kernel_launch(void* const* d_in, const int* in_sizes, int n_in,
                              void* d_out, int out_size, void* d_ws, size_t ws_size,
                              hipStream_t stream) {
  const int*   cat   = (const int*)d_in[0];
  const float* emb   = (const float*)d_in[1];
  const float* acc_w = (const float*)d_in[2];
  const float* acc_b = (const float*)d_in[3];
  const float* gw    = (const float*)d_in[4];
  const float* gb    = (const float*)d_in[5];
  const float* lw    = (const float*)d_in[6];
  const float* lb    = (const float*)d_in[7];
  float* out = (float*)d_out;

  const int B = in_sizes[0] / L_CATS;   // 4096
  const bool preconv = (ws_size >= (size_t)WBYTES);

  if (preconv) {
    _Float16* wsf = (_Float16*)d_ws;
    hipLaunchKernelGGL(prep_weights, dim3(48), dim3(256), 0, stream,
                       acc_w, gw, wsf);
    hipLaunchKernelGGL((catembed_fused<true>), dim3(B), dim3(256), 0, stream,
                       cat, emb, acc_w, acc_b, gw, gb, lw, lb,
                       (const _Float16*)wsf, out);
  } else {
    hipLaunchKernelGGL((catembed_fused<false>), dim3(B), dim3(256), 0, stream,
                       cat, emb, acc_w, acc_b, gw, gb, lw, lb,
                       (const _Float16*)nullptr, out);
  }
}
